// TransliterationModelattention_18210661335283
// MI455X (gfx1250) — matrix operand order repaired and checked by reading, not compiled, not run on hardware
//
#include <hip/hip_runtime.h>
#include <hip/hip_bf16.h>

// ---------------------------------------------------------------------------
// Problem constants (from the reference)
// ---------------------------------------------------------------------------
#define BB   1024   // batch
#define SS   48     // source length
#define TT   32     // target length
#define EE   256    // embedding dim
#define HH   512    // hidden dim
#define GG   (4*HH) // gate dim = 2048
#define VV   96     // vocab out

typedef __attribute__((ext_vector_type(16))) __bf16 v16bf;
typedef __attribute__((ext_vector_type(8)))  float  v8f;

// ---------------------------------------------------------------------------
// WMMA GEMM:  Out[M x N] = A[M x K] * W[N x K]^T + bias[N]
// A, W bf16 row-major (lda/ldw elements); Out f32 (ldo).
// Requirements (guaranteed by all call sites):
//   M % 128 == 0, K % 32 == 0, all A/W rows 16B-aligned (lda/ldw % 8 == 0,
//   base pointers 16B aligned). N may be ragged (W row index is clamped;
//   stores are guarded; clamped B columns never affect valid output columns).
// Block tile 128x128, 256 threads = 8 waves (2x4), each wave 64x32
// (4x2 wmma 16x16 tiles), K staged in chunks of 32 through LDS.
// ---------------------------------------------------------------------------
__global__ __launch_bounds__(256) void wmma_gemm(
    const __bf16* __restrict__ A, int lda,
    const __bf16* __restrict__ W, int ldw,
    const float* __restrict__ bias,
    float* __restrict__ Out, int ldo,
    int M, int N, int K)
{
    __shared__ __align__(16) __bf16 As[128 * 32];   // 8 KB
    __shared__ __align__(16) __bf16 Ws[128 * 32];   // 8 KB

    const int tid   = threadIdx.x;
    const int lane  = tid & 31;
    const int wave  = tid >> 5;
    const int half  = lane >> 4;          // 0 or 1
    const int l16   = lane & 15;
    const int waveM = (wave >> 2) * 64;   // 0 / 64
    const int waveN = (wave & 3)  * 32;   // 0 / 32 / 64 / 96
    const int blockM = blockIdx.y * 128;
    const int blockN = blockIdx.x * 128;

    v8f acc[4][2];
#pragma unroll
    for (int mi = 0; mi < 4; ++mi)
#pragma unroll
        for (int ni = 0; ni < 2; ++ni) {
            int col = blockN + waveN + ni * 16 + l16;
            float b = (bias != nullptr && col < N) ? bias[col] : 0.0f;
#pragma unroll
            for (int r = 0; r < 8; ++r) acc[mi][ni][r] = b;
        }

    for (int k0 = 0; k0 < K; k0 += 32) {
        __syncthreads();
        // Stage 128x32 bf16 tiles via unguarded 16B vector loads.
        // 512 uint4 per tile, 256 threads -> 2 per thread per tile.
#pragma unroll
        for (int i = 0; i < 2; ++i) {
            int p   = tid + i * 256;
            int row = p >> 2;            // 0..127
            int col = (p & 3) * 8;       // 0 / 8 / 16 / 24
            // A: M % 128 == 0 -> always in range.
            uint4 av = *reinterpret_cast<const uint4*>(
                A + (long)(blockM + row) * lda + k0 + col);
            *reinterpret_cast<uint4*>(&As[row * 32 + col]) = av;
            // W: clamp ragged N (harmless for valid output columns).
            int wr = blockN + row;
            wr = (wr < N) ? wr : (N - 1);
            uint4 wv = *reinterpret_cast<const uint4*>(
                W + (long)wr * ldw + k0 + col);
            *reinterpret_cast<uint4*>(&Ws[row * 32 + col]) = wv;
        }
        __syncthreads();

        // A fragment (16-bit A 16x32 layout): lane (half,l16) holds
        // K = half*8 + [0..7] and K = 16 + half*8 + [0..7] -> two 16B runs.
        v16bf afrag[4], bfrag[2];
#pragma unroll
        for (int mi = 0; mi < 4; ++mi) {
            const __bf16* base = &As[(waveM + mi * 16 + l16) * 32];
            uint4* d = reinterpret_cast<uint4*>(&afrag[mi]);
            d[0] = *reinterpret_cast<const uint4*>(base + half * 8);
            d[1] = *reinterpret_cast<const uint4*>(base + 16 + half * 8);
        }
        // B fragment (16-bit B 32x16): lane column n=l16 holds
        // K = half*8 + [0..7] and K = 16 + half*8 + [0..7] -> two 16B runs of W[n][k].
#pragma unroll
        for (int ni = 0; ni < 2; ++ni) {
            const __bf16* base = &Ws[(waveN + ni * 16 + l16) * 32];
            uint4* d = reinterpret_cast<uint4*>(&bfrag[ni]);
            d[0] = *reinterpret_cast<const uint4*>(base + half * 8);
            d[1] = *reinterpret_cast<const uint4*>(base + 16 + half * 8);
        }

#pragma unroll
        for (int mi = 0; mi < 4; ++mi)
#pragma unroll
            for (int ni = 0; ni < 2; ++ni)
                acc[mi][ni] = __builtin_amdgcn_wmma_f32_16x16x32_bf16(
                    false, afrag[mi], false, bfrag[ni],
                    (short)0, acc[mi][ni], false, false);
    }

    // C/D layout: lane col = l16, VGPR r -> row = half*8 + r.
#pragma unroll
    for (int mi = 0; mi < 4; ++mi)
#pragma unroll
        for (int ni = 0; ni < 2; ++ni) {
            int col = blockN + waveN + ni * 16 + l16;
            if (col >= N) continue;
#pragma unroll
            for (int r = 0; r < 8; ++r) {
                int row = blockM + waveM + mi * 16 + half * 8 + r;
                Out[(long)row * ldo + col] = acc[mi][ni][r];
            }
        }
}

// ---------------------------------------------------------------------------
// Elementwise / helper kernels
// ---------------------------------------------------------------------------
__global__ void fill_f32(float* __restrict__ p, float v, long n) {
    long i = (long)blockIdx.x * blockDim.x + threadIdx.x;
    if (i < n) p[i] = v;
}

__global__ void fill_bf16(__bf16* __restrict__ p, long n) {
    long i = (long)blockIdx.x * blockDim.x + threadIdx.x;
    if (i < n) p[i] = (__bf16)0.0f;
}

// dst[r, 0:wa+wb] = bf16(concat(Asrc[r,0:wa], Bsrc[r,0:wb]))
__global__ void concat2_bf16(const float* __restrict__ Asrc, int wa,
                             const float* __restrict__ Bsrc, int wb,
                             __bf16* __restrict__ dst, long rows) {
    int wt = wa + wb;
    long i = (long)blockIdx.x * blockDim.x + threadIdx.x;
    if (i >= rows * wt) return;
    long r = i / wt;
    int  c = (int)(i % wt);
    float v = (c < wa) ? Asrc[r * wa + c] : Bsrc[r * wb + (c - wa)];
    dst[i] = (__bf16)v;
}

// dst[r, 0:cols] = bf16(src[r*ld + off + c])
__global__ void slice_bf16(const float* __restrict__ src, int ld, int off,
                           __bf16* __restrict__ dst, long rows, int cols) {
    long i = (long)blockIdx.x * blockDim.x + threadIdx.x;
    if (i >= rows * cols) return;
    long r = i / cols;
    int  c = (int)(i % cols);
    dst[i] = (__bf16)src[r * ld + off + c];
}

// dst[b*dst_ld + e] = bf16(table[toks[b*tok_stride] * width + e])
__global__ void embed_bf16(const int* __restrict__ toks, int tok_stride,
                           const float* __restrict__ table, int width,
                           __bf16* __restrict__ dst, int dst_ld, int nrows) {
    long i = (long)blockIdx.x * blockDim.x + threadIdx.x;
    if (i >= (long)nrows * width) return;
    int b = (int)(i / width);
    int e = (int)(i % width);
    int tok = toks[(long)b * tok_stride];
    dst[(long)b * dst_ld + e] = (__bf16)table[(long)tok * width + e];
}

// LSTM cell: gates (B x 4H) f32 -> c update; h written as bf16 into the
// next-step GEMM A-buffer slot, optionally also into enc_out (bf16 + f32).
__global__ void lstm_cell(const float* __restrict__ g, float* __restrict__ c,
                          __bf16* __restrict__ hA, int hA_ld,
                          __bf16* __restrict__ hout_bf, int bf_ld,
                          float* __restrict__ hout_f, int f_ld) {
    int i = blockIdx.x * blockDim.x + threadIdx.x;
    if (i >= BB * HH) return;
    int b = i / HH, j = i % HH;
    const float* gb = g + (long)b * GG;
    float ig = gb[j], fg = gb[HH + j], gg = gb[2 * HH + j], og = gb[3 * HH + j];
    float si = 1.0f / (1.0f + expf(-ig));
    float sf = 1.0f / (1.0f + expf(-fg));
    float so = 1.0f / (1.0f + expf(-og));
    float cn = sf * c[i] + si * tanhf(gg);
    float hn = so * tanhf(cn);
    c[i] = cn;
    hA[(long)b * hA_ld + j] = (__bf16)hn;
    if (hout_bf) hout_bf[(long)b * bf_ld + j] = (__bf16)hn;
    if (hout_f)  hout_f[(long)b * f_ld + j]  = hn;
}

__global__ void copy_h_bf16(const __bf16* __restrict__ src, int sld,
                            __bf16* __restrict__ dst, int dld) {
    int i = blockIdx.x * blockDim.x + threadIdx.x;
    if (i >= BB * HH) return;
    int b = i / HH, j = i % HH;
    dst[(long)b * dld + j] = src[(long)b * sld + j];
}

// energy[b,s] = sum_k tanh(hp[b,k] + part[b,s,k]) * v[k]   (attn_b folded in part)
// one wave per (b,s)
__global__ void attn_energy(const float* __restrict__ hp,
                            const float* __restrict__ part,
                            const float* __restrict__ v,
                            float* __restrict__ energy) {
    int wid  = (int)(((long)blockIdx.x * blockDim.x + threadIdx.x) >> 5);
    int lane = threadIdx.x & 31;
    if (wid >= BB * SS) return;
    int b = wid / SS, s = wid % SS;
    const float* hpb = hp + (long)b * HH;
    const float* pb  = part + ((long)b * SS + s) * HH;
    float sum = 0.0f;
    for (int k = lane; k < HH; k += 32)
        sum += tanhf(hpb[k] + pb[k]) * v[k];
    for (int off = 16; off; off >>= 1)
        sum += __shfl_xor(sum, off, 32);
    if (lane == 0) energy[wid] = sum;
}

// softmax over S=48, one wave per batch row; writes aw and d_out attention slice
__global__ void attn_softmax(const float* __restrict__ energy,
                             float* __restrict__ aw,
                             float* __restrict__ attn_out) {
    int wid  = (int)(((long)blockIdx.x * blockDim.x + threadIdx.x) >> 5);
    int lane = threadIdx.x & 31;
    if (wid >= BB) return;
    const float* e = energy + (long)wid * SS;
    float e0 = e[lane];
    float e1 = (lane < SS - 32) ? e[32 + lane] : -1e30f;
    float m = fmaxf(e0, e1);
    for (int off = 16; off; off >>= 1) m = fmaxf(m, __shfl_xor(m, off, 32));
    float x0 = expf(e0 - m);
    float x1 = (lane < SS - 32) ? expf(e1 - m) : 0.0f;
    float s = x0 + x1;
    for (int off = 16; off; off >>= 1) s += __shfl_xor(s, off, 32);
    float inv = 1.0f / s;
    float* ao = attn_out + (long)wid * TT * SS;  // attn_out already offset by (t+1)*SS
    aw[(long)wid * SS + lane] = x0 * inv;
    ao[lane] = x0 * inv;
    if (lane < SS - 32) {
        aw[(long)wid * SS + 32 + lane] = x1 * inv;
        ao[32 + lane] = x1 * inv;
    }
}

// ctx[b,j] = sum_s aw[b,s] * enc_out[b,s,j]  -> bf16 into decoder A ctx slot
__global__ void attn_ctx(const float* __restrict__ aw,
                         const float* __restrict__ enc_out,
                         __bf16* __restrict__ dst, int dst_ld) {
    int i = blockIdx.x * blockDim.x + threadIdx.x;
    if (i >= BB * HH) return;
    int b = i / HH, j = i % HH;
    const float* awb = aw + (long)b * SS;
    const float* eb  = enc_out + (long)b * SS * HH + j;
    float sum = 0.0f;
#pragma unroll 4
    for (int s = 0; s < SS; ++s) sum += awb[s] * eb[(long)s * HH];
    dst[(long)b * dst_ld + j] = (__bf16)sum;
}

// ---------------------------------------------------------------------------
// Host orchestration
// ---------------------------------------------------------------------------
static inline long cdiv(long a, long b) { return (a + b - 1) / b; }

extern "C" void kernel_launch(void* const* d_in, const int* in_sizes, int n_in,
                              void* d_out, int out_size, void* d_ws, size_t ws_size,
                              hipStream_t stream) {
    const int*   source  = (const int*)d_in[0];
    const int*   target  = (const int*)d_in[1];
    const float* enc_emb = (const float*)d_in[2];
    const float* enc_Wih = (const float*)d_in[3];
    const float* enc_Whh = (const float*)d_in[4];
    const float* enc_b   = (const float*)d_in[5];
    const float* dec_emb = (const float*)d_in[6];
    const float* dec_Wih = (const float*)d_in[7];
    const float* dec_Whh = (const float*)d_in[8];
    const float* dec_b   = (const float*)d_in[9];
    const float* attn_W  = (const float*)d_in[10];
    const float* attn_b  = (const float*)d_in[11];
    const float* attn_v  = (const float*)d_in[12];
    const float* out_W   = (const float*)d_in[13];
    const float* out_b   = (const float*)d_in[14];
    float* out = (float*)d_out;

    // Bump allocator over workspace (all offsets 256B aligned).
    char* ws = (char*)d_ws;
    auto alloc = [&](size_t bytes) -> char* {
        char* p = ws;
        ws += (bytes + 255) & ~(size_t)255;
        return p;
    };
    __bf16* encW   = (__bf16*)alloc((size_t)GG * (EE + HH) * 2);      // 2048 x 768
    __bf16* decW   = (__bf16*)alloc((size_t)GG * (EE + 2 * HH) * 2);  // 2048 x 1280
    __bf16* WaH    = (__bf16*)alloc((size_t)HH * HH * 2);             // 512 x 512
    __bf16* WaE    = (__bf16*)alloc((size_t)HH * HH * 2);             // 512 x 512
    __bf16* outWb  = (__bf16*)alloc((size_t)VV * HH * 2);             // 96 x 512
    __bf16* Aenc   = (__bf16*)alloc((size_t)BB * (EE + HH) * 2);      // [x | h]
    __bf16* Adec   = (__bf16*)alloc((size_t)BB * (EE + 2 * HH) * 2);  // [emb | ctx | h]
    float*  gates  = (float*)alloc((size_t)BB * GG * 4);
    float*  cbuf   = (float*)alloc((size_t)BB * HH * 4);
    float*  encOutF = (float*)alloc((size_t)BB * SS * HH * 4);
    __bf16* encOutB = (__bf16*)alloc((size_t)BB * SS * HH * 2);
    float*  encPart = (float*)alloc((size_t)BB * SS * HH * 4);        // + attn_b folded
    float*  hp     = (float*)alloc((size_t)BB * HH * 4);
    float*  energy = (float*)alloc((size_t)BB * SS * 4);
    float*  aw     = (float*)alloc((size_t)BB * SS * 4);

    const int KE = EE + HH;        // 768
    const int KD = EE + 2 * HH;    // 1280

    // ---- one-time prep (re-run every call: deterministic) ----
    fill_f32<<<cdiv((long)out_size, 256), 256, 0, stream>>>(out, 0.0f, out_size);
    fill_f32<<<cdiv((long)BB * HH, 256), 256, 0, stream>>>(cbuf, 0.0f, (long)BB * HH);
    fill_bf16<<<cdiv((long)BB * KE, 256), 256, 0, stream>>>(Aenc, (long)BB * KE);   // h0 = 0

    concat2_bf16<<<cdiv((long)GG * KE, 256), 256, 0, stream>>>(enc_Wih, EE, enc_Whh, HH, encW, GG);
    concat2_bf16<<<cdiv((long)GG * KD, 256), 256, 0, stream>>>(dec_Wih, EE + HH, dec_Whh, HH, decW, GG);
    slice_bf16<<<cdiv((long)HH * HH, 256), 256, 0, stream>>>(attn_W, 2 * HH, 0,  WaH, HH, HH);
    slice_bf16<<<cdiv((long)HH * HH, 256), 256, 0, stream>>>(attn_W, 2 * HH, HH, WaE, HH, HH);
    slice_bf16<<<cdiv((long)VV * HH, 256), 256, 0, stream>>>(out_W, HH, 0, outWb, VV, HH);

    // ---- encoder: 48 sequential LSTM steps ----
    dim3 gemmGatesGrid(GG / 128, BB / 128);            // 16 x 8
    for (int t = 0; t < SS; ++t) {
        embed_bf16<<<cdiv((long)BB * EE, 256), 256, 0, stream>>>(
            source + t, SS, enc_emb, EE, Aenc, KE, BB);
        wmma_gemm<<<gemmGatesGrid, 256, 0, stream>>>(
            Aenc, KE, encW, KE, enc_b, gates, GG, BB, GG, KE);
        lstm_cell<<<cdiv((long)BB * HH, 256), 256, 0, stream>>>(
            gates, cbuf, Aenc + EE, KE,
            encOutB + (long)t * HH, SS * HH,
            encOutF + (long)t * HH, SS * HH);
    }
    // hT -> decoder A h-slot
    copy_h_bf16<<<cdiv((long)BB * HH, 256), 256, 0, stream>>>(Aenc + EE, KE, Adec + EE + HH, KD);

    // ---- enc_part = enc_out @ Wa_e^T + attn_b   ((B*S) x 512 x 512) ----
    {
        dim3 g(HH / 128, (BB * SS) / 128);             // 4 x 384
        wmma_gemm<<<g, 256, 0, stream>>>(encOutB, HH, WaE, HH, attn_b,
                                         encPart, HH, BB * SS, HH, HH);
    }

    // ---- decoder: 31 sequential steps ----
    float* attnBase = out + (long)BB * TT * VV;   // attentions after predictions
    dim3 hpGrid(HH / 128, BB / 128);               // 4 x 8
    dim3 outGrid((VV + 127) / 128, BB / 128);      // 1 x 8
    for (int t = 0; t < TT - 1; ++t) {
        embed_bf16<<<cdiv((long)BB * EE, 256), 256, 0, stream>>>(
            target + t, TT, dec_emb, EE, Adec, KD, BB);
        // hp = h @ Wa_h^T
        wmma_gemm<<<hpGrid, 256, 0, stream>>>(
            Adec + EE + HH, KD, WaH, HH, nullptr, hp, HH, BB, HH, HH);
        attn_energy<<<(BB * SS) / 8, 256, 0, stream>>>(hp, encPart, attn_v, energy);
        attn_softmax<<<BB / 8, 256, 0, stream>>>(energy, aw, attnBase + (long)(t + 1) * SS);
        attn_ctx<<<cdiv((long)BB * HH, 256), 256, 0, stream>>>(aw, encOutF, Adec + EE, KD);
        // gates = [emb|ctx|h] @ dec_W^T + dec_b
        wmma_gemm<<<gemmGatesGrid, 256, 0, stream>>>(
            Adec, KD, decW, KD, dec_b, gates, GG, BB, GG, KD);
        lstm_cell<<<cdiv((long)BB * HH, 256), 256, 0, stream>>>(
            gates, cbuf, Adec + EE + HH, KD, nullptr, 0, nullptr, 0);
        // predictions[:, t+1, :] = h @ out_W^T + out_b
        wmma_gemm<<<outGrid, 256, 0, stream>>>(
            Adec + EE + HH, KD, outWb, HH, out_b,
            out + (long)(t + 1) * VV, TT * VV, BB, VV, HH);
    }
}
